// SpatialMultiHeadAttention_24283745092065
// MI455X (gfx1250) — matrix operand order repaired and checked by reading, not compiled, not run on hardware
//
#include <hip/hip_runtime.h>
#include <hip/hip_bf16.h>

// CDNA5 / gfx1250, wave32. WMMA f16 16x16x32 with f32 accumulate everywhere.

typedef __attribute__((ext_vector_type(16))) _Float16 v16h;
typedef __attribute__((ext_vector_type(8)))  float    v8f;
typedef __attribute__((ext_vector_type(2)))  _Float16 h2;    // LDS f16 pairs
typedef __attribute__((ext_vector_type(2)))  __fp16   cvt2;  // cvt_pkrtz result type

#define D_MODEL    512
#define N_HEADS    8
#define D_K        64
#define SEQ        1024
#define BATCH      2

// ---- WMMA f16 fragment index helpers (per CDNA5 ISA layouts) ----
// A 16x32 f16: lane = {hi,m}; VGPR v holds K = kA, kA+1
__device__ __forceinline__ int kA16(int v, int hi) {
    return ((v & 3) << 1) + (hi << 3) + ((v >> 2) << 4);
}
// B 32x16 f16: lane = {hi,n}; VGPR v holds K = kB, kB+1 (mirrors 64x16 table)
__device__ __forceinline__ int kB16(int v, int hi) {
    return ((v & 3) << 1) + (hi << 3) + ((v >> 2) << 4);
}
// C/D 16x16 f32: VGPR v -> (m = v + 8*hi, n = lane&15)

#define WMMA_F16(a, b, c) \
    __builtin_amdgcn_wmma_f32_16x16x32_f16(false, (a), false, (b), (short)0, (c), false, false)

// ---------------------------------------------------------------------------
// GEMM: C[M,512] = A[M,512] @ W[512,512] + bias[512]
// Block: 256 threads = 8 waves; block tile 64(M) x 64(N); K-step 32.
// Wave w: wm = w&3 (16-row slab), wn = w>>2 (32-col slab -> 2 WMMAs/k-step).
// ---------------------------------------------------------------------------
__global__ __launch_bounds__(256) void gemm512_kernel(
    const float* __restrict__ A, const float* __restrict__ W,
    const float* __restrict__ bias, float* __restrict__ C, int M)
{
    const int KD = D_MODEL, ND = D_MODEL;
    __shared__ _Float16 As[64][40];   // padded rows: no bank conflicts
    __shared__ _Float16 Bs[64][40];   // W tile stored transposed: Bs[n][k]

    const int tid  = threadIdx.x;
    const int lane = tid & 31;
    const int wave = tid >> 5;
    const int hi   = (lane >> 4) & 1, l15 = lane & 15;
    const int wm   = wave & 3;        // 4 m-tiles of 16
    const int wn   = wave >> 2;       // 2 n-slabs of 32
    const int m0   = blockIdx.y * 64;
    const int n0   = blockIdx.x * 64;

    // fixed staging coordinates (fully unrolled, float4 per transaction)
    const int ar = tid >> 3;          // A rows ar, ar+32
    const int ak = (tid & 7) * 4;     // A k-quad
    const int bk = tid >> 4;          // W k-rows bk, bk+16
    const int bn = (tid & 15) * 4;    // W n-quad

    v8f acc0 = {}, acc1 = {};
    for (int k0 = 0; k0 < KD; k0 += 32) {
        #pragma unroll
        for (int t = 0; t < 2; ++t) {   // A: 64x32 tile
            int r = ar + t * 32;
            float4 a4 = *(const float4*)&A[(size_t)(m0 + r) * KD + k0 + ak];
            cvt2 p0 = __builtin_amdgcn_cvt_pkrtz(a4.x, a4.y);
            cvt2 p1 = __builtin_amdgcn_cvt_pkrtz(a4.z, a4.w);
            As[r][ak + 0] = (_Float16)p0[0]; As[r][ak + 1] = (_Float16)p0[1];
            As[r][ak + 2] = (_Float16)p1[0]; As[r][ak + 3] = (_Float16)p1[1];
        }
        #pragma unroll
        for (int t = 0; t < 2; ++t) {   // W: 32x64 tile, transposed into Bs[n][k]
            int k = bk + t * 16;
            float4 w4 = *(const float4*)&W[(size_t)(k0 + k) * ND + n0 + bn];
            cvt2 p0 = __builtin_amdgcn_cvt_pkrtz(w4.x, w4.y);
            cvt2 p1 = __builtin_amdgcn_cvt_pkrtz(w4.z, w4.w);
            Bs[bn + 0][k] = (_Float16)p0[0]; Bs[bn + 1][k] = (_Float16)p0[1];
            Bs[bn + 2][k] = (_Float16)p1[0]; Bs[bn + 3][k] = (_Float16)p1[1];
        }
        if (k0 + 32 < KD) {             // pull next slab toward L2/WGP$
            __builtin_prefetch(&A[(size_t)(m0 + ar) * KD + k0 + 32 + ak], 0, 1);
            __builtin_prefetch(&W[(size_t)(k0 + 32 + bk) * ND + n0 + bn], 0, 1);
        }
        __syncthreads();

        v16h af, bf0, bf1;
        #pragma unroll
        for (int v = 0; v < 8; ++v) {
            int ka = kA16(v, hi);
            h2 a = *(const h2*)&As[wm * 16 + l15][ka];
            af[2 * v] = a[0]; af[2 * v + 1] = a[1];
            int kb = kB16(v, hi);
            h2 b0 = *(const h2*)&Bs[wn * 32 + l15][kb];
            h2 b1 = *(const h2*)&Bs[wn * 32 + 16 + l15][kb];
            bf0[2 * v] = b0[0]; bf0[2 * v + 1] = b0[1];
            bf1[2 * v] = b1[0]; bf1[2 * v + 1] = b1[1];
        }
        acc0 = WMMA_F16(af, bf0, acc0);
        acc1 = WMMA_F16(af, bf1, acc1);
        __syncthreads();
    }

    #pragma unroll
    for (int v = 0; v < 8; ++v) {
        int m = m0 + wm * 16 + v + hi * 8;
        int n = n0 + wn * 32 + l15;
        C[(size_t)m * ND + n]      = acc0[v] + bias[n];
        C[(size_t)m * ND + n + 16] = acc1[v] + bias[n + 16];
    }
}

// ---------------------------------------------------------------------------
// Fused attention per (b,h,q-tile of 16):
//   scores = (Q Kt)/8 + MLP(dist)  -> LDS (16 x 1024 f32)
//   softmax rows (one wave per row) -> f16 probs in LDS
//   out = P @ V  -> ctx (laid out (b*S+s, h*64+d) for the O-projection GEMM)
// Block: 512 threads = 16 waves. No (S x S) materialization in HBM.
// ---------------------------------------------------------------------------
#define SCP (SEQ + 4)   // f32 score row stride (bank-conflict pad)
#define SPP (SEQ + 8)   // f16 prob  row stride (bank-conflict pad)

__global__ __launch_bounds__(512) void attn_kernel(
    const float* __restrict__ Qp, const float* __restrict__ Kp,
    const float* __restrict__ Vp, const float* __restrict__ dist,
    const float* __restrict__ W1, const float* __restrict__ b1,
    const float* __restrict__ W2, const float* __restrict__ b2,
    float* __restrict__ ctx)
{
    extern __shared__ char smem[];
    float*    sc = (float*)smem;                                 // [16][SCP]
    _Float16* sp = (_Float16*)(smem + 16 * SCP * sizeof(float)); // [16][SPP]
    __shared__ float w1s[64], b1s[64], w2s[64];                  // MLP params (col h)
    __shared__ float b2h_s;

    const int tid  = threadIdx.x;
    const int lane = tid & 31, wave = tid >> 5;
    const int hi   = (lane >> 4) & 1, l15 = lane & 15;
    const int bh = blockIdx.y, bb = bh >> 3, h = bh & 7;
    const int q0 = blockIdx.x * 16;

    if (tid < 64) {
        w1s[tid] = W1[tid];
        b1s[tid] = b1[tid];
        w2s[tid] = W2[tid * N_HEADS + h];
    }
    if (tid == 0) b2h_s = b2[h];
    __syncthreads();

    // Q fragments for this q-tile (dk=64 -> two K32 halves), built once per wave
    const float* Qrow = Qp + ((size_t)(bb * SEQ + q0 + l15)) * D_MODEL + h * D_K;
    v16h aq0, aq1;
    #pragma unroll
    for (int v = 0; v < 8; ++v) {
        int ka = kA16(v, hi);
        float2 q2 = *(const float2*)&Qrow[ka];
        float2 q3 = *(const float2*)&Qrow[32 + ka];
        cvt2 p0 = __builtin_amdgcn_cvt_pkrtz(q2.x, q2.y);
        cvt2 p1 = __builtin_amdgcn_cvt_pkrtz(q3.x, q3.y);
        aq0[2 * v] = (_Float16)p0[0]; aq0[2 * v + 1] = (_Float16)p0[1];
        aq1[2 * v] = (_Float16)p1[0]; aq1[2 * v + 1] = (_Float16)p1[1];
    }

    const float scale = 0.125f; // 1/sqrt(dk)

    // ---- Phase 1: scores + spatial-bias MLP; 64 j-tiles over 16 waves ----
    for (int t = 0; t < 4; ++t) {
        const int j0 = (wave + t * 16) * 16;
        const float* Krow = Kp + ((size_t)(bb * SEQ + j0 + l15)) * D_MODEL + h * D_K;
        v16h bk0, bk1;
        #pragma unroll
        for (int v = 0; v < 8; ++v) {
            int kb = kB16(v, hi);
            float2 k2v = *(const float2*)&Krow[kb];
            float2 k3v = *(const float2*)&Krow[32 + kb];
            cvt2 p0 = __builtin_amdgcn_cvt_pkrtz(k2v.x, k2v.y);
            cvt2 p1 = __builtin_amdgcn_cvt_pkrtz(k3v.x, k3v.y);
            bk0[2 * v] = (_Float16)p0[0]; bk0[2 * v + 1] = (_Float16)p0[1];
            bk1[2 * v] = (_Float16)p1[0]; bk1[2 * v + 1] = (_Float16)p1[1];
        }
        v8f c = {};
        c = WMMA_F16(aq0, bk0, c);
        c = WMMA_F16(aq1, bk1, c);

        // spatial-bias MLP: k2 outer so 3 LDS reads feed 8 rows of FMA work
        float dv[8], gv[8];
        #pragma unroll
        for (int v = 0; v < 8; ++v) {
            int m = v + hi * 8;
            dv[v] = dist[((size_t)bb * SEQ + q0 + m) * SEQ + j0 + l15];
            gv[v] = b2h_s;
        }
        for (int k2 = 0; k2 < 64; ++k2) {
            float w1k = w1s[k2], b1k = b1s[k2], w2k = w2s[k2];
            #pragma unroll
            for (int v = 0; v < 8; ++v) {
                float hk = fmaxf(fmaf(dv[v], w1k, b1k), 0.f);
                gv[v] = fmaf(hk, w2k, gv[v]);
            }
        }
        #pragma unroll
        for (int v = 0; v < 8; ++v) {
            int m = v + hi * 8;
            sc[m * SCP + j0 + l15] = c[v] * scale + gv[v];
        }
    }
    __syncthreads();

    // ---- Phase 2: softmax, one wave per row; write f16 probs ----
    {
        float* row = sc + wave * SCP;
        float mx = -3.4e38f;
        for (int j = lane; j < SEQ; j += 32) mx = fmaxf(mx, row[j]);
        #pragma unroll
        for (int o = 16; o > 0; o >>= 1) mx = fmaxf(mx, __shfl_xor(mx, o, 32));
        float sum = 0.f;
        for (int j = lane; j < SEQ; j += 32) {
            float e = __expf(row[j] - mx);
            row[j] = e;
            sum += e;
        }
        #pragma unroll
        for (int o = 16; o > 0; o >>= 1) sum += __shfl_xor(sum, o, 32);
        float inv = 1.f / sum;
        _Float16* prow = sp + wave * SPP;
        for (int j = lane; j < SEQ; j += 32) prow[j] = (_Float16)(row[j] * inv);
    }
    __syncthreads();

    // ---- Phase 3: out(16x64) = P(16x1024) @ V(1024x64); 4 waves, 16 cols each ----
    if (wave < 4) {
        const int nt = wave;
        v8f acc = {};
        for (int j0 = 0; j0 < SEQ; j0 += 32) {
            v16h ap, bv;
            #pragma unroll
            for (int v = 0; v < 8; ++v) {
                int ka = kA16(v, hi);
                h2 pr = *(const h2*)&sp[l15 * SPP + j0 + ka];
                ap[2 * v] = pr[0]; ap[2 * v + 1] = pr[1];
                int kb = kB16(v, hi);
                const float* Vr = Vp + ((size_t)(bb * SEQ + j0 + kb)) * D_MODEL
                                  + h * D_K + nt * 16 + l15;
                cvt2 pv = __builtin_amdgcn_cvt_pkrtz(Vr[0], Vr[D_MODEL]);
                bv[2 * v] = (_Float16)pv[0]; bv[2 * v + 1] = (_Float16)pv[1];
            }
            acc = WMMA_F16(ap, bv, acc);
        }
        #pragma unroll
        for (int v = 0; v < 8; ++v) {
            int m = v + hi * 8;
            ctx[((size_t)(bb * SEQ + q0 + m)) * D_MODEL + h * D_K + nt * 16 + l15] = acc[v];
        }
    }
}

// ---------------------------------------------------------------------------
extern "C" void kernel_launch(void* const* d_in, const int* in_sizes, int n_in,
                              void* d_out, int out_size, void* d_ws, size_t ws_size,
                              hipStream_t stream) {
    const float* x    = (const float*)d_in[0];
    const float* dist = (const float*)d_in[1];
    const float* Wq = (const float*)d_in[2];  const float* bq = (const float*)d_in[3];
    const float* Wk = (const float*)d_in[4];  const float* bk = (const float*)d_in[5];
    const float* Wv = (const float*)d_in[6];  const float* bv = (const float*)d_in[7];
    const float* Wo = (const float*)d_in[8];  const float* bo = (const float*)d_in[9];
    const float* W1 = (const float*)d_in[10]; const float* b1 = (const float*)d_in[11];
    const float* W2 = (const float*)d_in[12]; const float* b2 = (const float*)d_in[13];
    float* out = (float*)d_out;

    const int M = BATCH * SEQ;                       // 2048 rows
    const size_t matElems = (size_t)M * D_MODEL;     // 1M floats = 4 MB
    float* Qp  = (float*)d_ws;                       // ws usage: 16 MB total
    float* Kp  = Qp + matElems;
    float* Vp  = Kp + matElems;
    float* ctx = Vp + matElems;

    dim3 gblk(256), ggrd(D_MODEL / 64, M / 64);      // 8 x 32 blocks
    gemm512_kernel<<<ggrd, gblk, 0, stream>>>(x, Wq, bq, Qp, M);
    gemm512_kernel<<<ggrd, gblk, 0, stream>>>(x, Wk, bk, Kp, M);
    gemm512_kernel<<<ggrd, gblk, 0, stream>>>(x, Wv, bv, Vp, M);

    dim3 agrd(SEQ / 16, BATCH * N_HEADS);            // 64 x 16 blocks
    size_t shmem = 16 * SCP * sizeof(float) + 16 * SPP * sizeof(_Float16); // ~99 KB (<320 KB/WGP)
    attn_kernel<<<agrd, dim3(512), shmem, stream>>>(Qp, Kp, Vp, dist, W1, b1, W2, b2, ctx);

    gemm512_kernel<<<ggrd, gblk, 0, stream>>>(ctx, Wo, bo, out, M);
}
